// Decoder_46943992546168
// MI455X (gfx1250) — matrix operand order repaired and checked by reading, not compiled, not run on hardware
//
#include <hip/hip_runtime.h>

// ---------------------------------------------------------------------------
// Show-Attend-Tell decoder for MI455X (gfx1250), wave32 + WMMA bf16 + TDM.
// B=128, L=196, T=20 (19 steps), A=E=N=D=512, V=10000.
// GEMMs: v_wmma_f32_16x16x32_bf16, f32 accumulate; A tile staged to LDS via
// tensor_load_to_lds (TDM), 32Mx64N per wave (8 WMMA / 32-k step).
// ---------------------------------------------------------------------------

#define BB   128
#define LL   196
#define TT   20
#define TM1  19
#define DD   512
#define VV   10000

typedef __attribute__((ext_vector_type(16))) __bf16 v16bf;
typedef __attribute__((ext_vector_type(8)))  float  v8f;
typedef __attribute__((ext_vector_type(4)))  unsigned int u32x4;
typedef __attribute__((ext_vector_type(8)))  int i32x8;
typedef __attribute__((ext_vector_type(4)))  int i32x4;

__device__ __forceinline__ unsigned short f2bf(float f) {
  union { float f; unsigned u; } v; v.f = f;
  unsigned r = v.u + 0x7FFFu + ((v.u >> 16) & 1u);   // round-to-nearest-even
  return (unsigned short)(r >> 16);
}
__device__ __forceinline__ float sigmoidf_(float x) { return 1.f / (1.f + __expf(-x)); }

// B 32x16 (KxN), B[k][n] = W[n][k]: lane n = lane&15; K = g*8..g*8+7 and 16+g*8..16+g*8+7 in W row n
__device__ __forceinline__ v16bf load_b_frag(const unsigned short* __restrict__ W,
                                             int ldw, int n0, int k, int lane) {
  size_t row = (size_t)(n0 + (lane & 15)) * (size_t)ldw;
  int g = lane >> 4;
  const unsigned short* p = W + row + k + g * 8;
  union { uint4 u[2]; v16bf v; } r;
  r.u[0] = *(const uint4*)(p);
  r.u[1] = *(const uint4*)(p + 16);
  return r.v;
}

// ---- generic bf16 WMMA GEMM: C = act(A(MxK) * W(NxK)^T + bias + add) ------
// block(32,4): 4 waves. Block tile 32M x 256N; wave tile 32M x 64N.
// A tile (32 x K bf16) staged to LDS by wave 0 via TDM tensor_load_to_lds.
// Requires M % 32 == 0, K % 32 == 0. grid((N+255)/256, M/32).
// act: 0 = none, 1 = tanh. outf (f32) and/or outb (bf16) may be null.
__global__ void __launch_bounds__(128)
gemm_bf16_kernel(const unsigned short* __restrict__ A, int lda,
                 const unsigned short* __restrict__ W, int ldw,
                 const float* __restrict__ bias,
                 const float* __restrict__ add, int lda_add,
                 float* __restrict__ outf, int ldf,
                 unsigned short* __restrict__ outb, int ldb,
                 int M, int N, int K, int act) {
  extern __shared__ unsigned short smem[];          // 32 x K bf16 A tile
  const int lane = threadIdx.x;
  const int wv   = threadIdx.y;
  const int m0   = blockIdx.y * 32;
  // wave-uniform N base / tile count, pinned to SGPRs so tile guards become
  // scalar branches (EXEC untouched around v_wmma)
  const int n0   = __builtin_amdgcn_readfirstlane(blockIdx.x * 256 + wv * 64);
  const bool active = n0 < N;

  if (wv == 0) {
    // ---- TDM: DMA the 32 x K A tile (row-major, stride lda) into LDS ----
    unsigned long long ga = (unsigned long long)(const void*)(A + (size_t)m0 * lda);
    int rows = M - m0;                              // rows available (>= 32)
    u32x4 g0;
    g0.x = 1u;                                      // count=1, user descriptor
    g0.y = 0u;                                      // lds_addr: dynamic-LDS base (no static LDS)
    g0.z = (unsigned)(ga & 0xFFFFFFFFull);          // global_addr[31:0]
    g0.w = (unsigned)((ga >> 32) & 0x01FFFFFFull) | (2u << 30);  // addr[56:32] | type=2
    i32x8 g1;
    g1[0] = 0x00010000;                             // data_size=1 (2B), wg_mask=0
    g1[1] = (int)((unsigned)(K & 0xFFFF) << 16);    // tensor_dim0[15:0] @ bits 63:48
    g1[2] = (int)(((unsigned)(rows & 0xFFFF) << 16) | ((unsigned)K >> 16)); // dim0 hi | dim1 lo
    g1[3] = (int)(((unsigned)(K & 0xFFFF) << 16) | ((unsigned)rows >> 16)); // tile_dim0=K | dim1 hi
    g1[4] = 32;                                     // tile_dim1 = 32 rows (tile_dim2 = 0)
    g1[5] = lda;                                    // tensor_dim0_stride[31:0] (elements)
    g1[6] = 0;
    g1[7] = 0;
    i32x4 gz4 = {0, 0, 0, 0};
    i32x8 gz8 = {0, 0, 0, 0, 0, 0, 0, 0};
    __builtin_amdgcn_tensor_load_to_lds(g0, g1, gz4, gz4, gz8, 0);
    __builtin_amdgcn_s_wait_tensorcnt(0);
  }
  __syncthreads();
  if (!active) return;                              // whole-wave uniform exit after barrier

  int nt = (N - n0) >> 4; if (nt > 4) nt = 4;       // valid 16-wide N tiles (1..4)
  nt = __builtin_amdgcn_readfirstlane(nt);          // SGPR: scalar guards below
  v8f acc0[4] = {};
  v8f acc1[4] = {};
  const int laneM = lane & 15;
  const int g8    = (lane >> 4) * 8;

  for (int k = 0; k < K; k += 32) {
    // A frags from LDS (layout identical to row-major global tile)
    union { uint4 u[2]; v16bf v; } A0, A1;
    int base0 = laneM * K + k + g8;
    A0.u[0] = *(const uint4*)&smem[base0];
    A0.u[1] = *(const uint4*)&smem[base0 + 16];
    int base1 = base0 + 16 * K;
    A1.u[0] = *(const uint4*)&smem[base1];
    A1.u[1] = *(const uint4*)&smem[base1 + 16];
#pragma unroll
    for (int j = 0; j < 4; ++j) {
      if (j < nt) {                                 // scalar (SGPR) guard: EXEC stays all-1s
        v16bf b = load_b_frag(W, ldw, n0 + j * 16, k, lane);
        acc0[j] = __builtin_amdgcn_wmma_f32_16x16x32_bf16(false, A0.v, false, b,
                                                          (short)0, acc0[j], false, false);
        acc1[j] = __builtin_amdgcn_wmma_f32_16x16x32_bf16(false, A1.v, false, b,
                                                          (short)0, acc1[j], false, false);
      }
    }
  }

  const int mhalf = (lane >> 4) ? 8 : 0;
#pragma unroll
  for (int j = 0; j < 4; ++j) {
    if (j < nt) {
      int n = n0 + j * 16 + laneM;
      float bv = bias ? bias[n] : 0.f;
#pragma unroll
      for (int half = 0; half < 2; ++half) {
        v8f accv = half ? acc1[j] : acc0[j];
        int mb = m0 + half * 16 + mhalf;
#pragma unroll
        for (int r = 0; r < 8; ++r) {
          int m = mb + r;
          float v = accv[r] + bv;
          if (add)  v += add[(size_t)m * lda_add + n];
          if (act == 1) v = tanhf(v);
          if (outf) outf[(size_t)m * ldf + n] = v;
          if (outb) outb[(size_t)m * ldb + n] = f2bf(v);
        }
      }
    }
  }
}

// ---- prep kernels ---------------------------------------------------------
__global__ void cvt_kernel(const float* __restrict__ src, unsigned short* __restrict__ dst,
                           size_t n) {
  size_t i = (size_t)blockIdx.x * blockDim.x + threadIdx.x;
  size_t s = (size_t)gridDim.x * blockDim.x;
  for (; i < n; i += s) dst[i] = f2bf(src[i]);
}
// dst row r = [A[r,0..ka) | Bm[r,0..kb)] in bf16
__global__ void pack2_kernel(const float* __restrict__ A, int ka,
                             const float* __restrict__ Bm, int kb,
                             unsigned short* __restrict__ dst) {
  int r = blockIdx.x, K = ka + kb;
  for (int c = threadIdx.x; c < K; c += blockDim.x) {
    float v = (c < ka) ? A[(size_t)r * ka + c] : Bm[(size_t)r * kb + (c - ka)];
    dst[(size_t)r * K + c] = f2bf(v);
  }
}
__global__ void addvec_kernel(const float* a, const float* b, float* o, int n) {
  int i = blockIdx.x * blockDim.x + threadIdx.x;
  if (i < n) o[i] = a[i] + b[i];
}
// mean over L annotation vectors -> bf16 (feeds h0/c0 GEMMs)
__global__ void mean_kernel(const float* __restrict__ enc, unsigned short* __restrict__ mean_b) {
  int b = blockIdx.x, d = threadIdx.x;
  float s = 0.f;
  for (int l = 0; l < LL; ++l) s += enc[((size_t)b * LL + l) * DD + d];
  mean_b[(size_t)b * DD + d] = f2bf(s * (1.f / (float)LL));
}

// ---- per-step kernels -----------------------------------------------------
__global__ void emb_kernel(const float* __restrict__ emb, const int* __restrict__ captions,
                           float* __restrict__ embf, unsigned short* __restrict__ xfull, int t) {
  int b = blockIdx.x, j = threadIdx.x;
  int tok = captions[b * TT + t];
  float v = emb[(size_t)tok * DD + j];
  embf[(size_t)b * DD + j] = v;
  xfull[(size_t)b * 1536 + j] = f2bf(v);
}

__global__ void beta_kernel(const float* __restrict__ h, const float* __restrict__ Wb,
                            const float* __restrict__ bb, float* __restrict__ beta) {
  int b = blockIdx.x * blockDim.x + threadIdx.x;
  if (b >= BB) return;
  float s = 0.f;
  for (int j = 0; j < DD; ++j) s += h[(size_t)b * DD + j] * Wb[j];
  beta[b] = sigmoidf_(s + bb[0]);
}

// one block per batch item: scores -> softmax -> context z (scaled by beta)
__global__ void __launch_bounds__(256)
attention_kernel(const float* __restrict__ enc_att, const float* __restrict__ dproj,
                 const float* __restrict__ enc, const float* __restrict__ beta,
                 const float* __restrict__ Watt, const float* __restrict__ batt,
                 const int* __restrict__ caplen, float* __restrict__ alphas_out,
                 float* __restrict__ z, unsigned short* __restrict__ xfull,
                 unsigned short* __restrict__ hz, int t) {
  int b = blockIdx.x, tid = threadIdx.x;
  int wave = tid >> 5, lane = tid & 31;
  __shared__ float s_dp[DD];
  __shared__ float s_e[LL];
  __shared__ float s_red[8];
  for (int i = tid; i < DD; i += 256) s_dp[i] = dproj[(size_t)b * DD + i];
  __syncthreads();
  // e_l = sum_a relu(enc_att + dproj) * W_att  (one wave per l, lanes over a)
  for (int l = wave; l < LL; l += 8) {
    const float* row = enc_att + ((size_t)b * LL + l) * DD;
    float p = 0.f;
    for (int a = lane; a < DD; a += 32) {
      float v = row[a] + s_dp[a];
      v = v > 0.f ? v : 0.f;
      p += v * Watt[a];
    }
    for (int off = 16; off > 0; off >>= 1) p += __shfl_xor(p, off, 32);
    if (lane == 0) s_e[l] = p + batt[0];
  }
  __syncthreads();
  // softmax over L
  float mymax = -1e30f;
  for (int l = tid; l < LL; l += 256) mymax = fmaxf(mymax, s_e[l]);
  for (int off = 16; off > 0; off >>= 1) mymax = fmaxf(mymax, __shfl_xor(mymax, off, 32));
  if (lane == 0) s_red[wave] = mymax;
  __syncthreads();
  float gmax = s_red[0];
  for (int w = 1; w < 8; ++w) gmax = fmaxf(gmax, s_red[w]);
  __syncthreads();
  float mysum = 0.f;
  for (int l = tid; l < LL; l += 256) { float ev = __expf(s_e[l] - gmax); s_e[l] = ev; mysum += ev; }
  for (int off = 16; off > 0; off >>= 1) mysum += __shfl_xor(mysum, off, 32);
  if (lane == 0) s_red[wave] = mysum;
  __syncthreads();
  float gsum = 0.f;
  for (int w = 0; w < 8; ++w) gsum += s_red[w];
  float inv = 1.f / gsum;
  __syncthreads();
  for (int l = tid; l < LL; l += 256) s_e[l] *= inv;
  __syncthreads();
  // masked alpha output
  int dl = caplen[b] - 1; dl = dl < 1 ? 1 : (dl > TM1 ? TM1 : dl);
  float mask = (t < dl) ? 1.f : 0.f;
  for (int l = tid; l < LL; l += 256)
    alphas_out[((size_t)b * TM1 + t) * LL + l] = s_e[l] * mask;
  // z = beta * sum_l alpha_l * enc[b,l,:]
  float bb2 = beta[b];
  for (int d = tid; d < DD; d += 256) {
    float acc = 0.f;
    const float* e0 = enc + (size_t)b * LL * DD + d;
    for (int l = 0; l < LL; ++l) acc += s_e[l] * e0[(size_t)l * DD];
    acc *= bb2;
    z[(size_t)b * DD + d] = acc;
    unsigned short bfv = f2bf(acc);
    xfull[(size_t)b * 1536 + DD + d] = bfv;   // z slot of [emb|z|h]
    hz[(size_t)b * 1024 + DD + d]    = bfv;   // z slot of [h2|z]
  }
}

__global__ void lstm_kernel(const float* __restrict__ gates, float* __restrict__ h,
                            float* __restrict__ c, unsigned short* __restrict__ xfull,
                            unsigned short* __restrict__ hz) {
  int b = blockIdx.x, j = threadIdx.x;
  const float* g = gates + (size_t)b * 2048;
  float ig = sigmoidf_(g[j]);
  float fg = sigmoidf_(g[512 + j]);
  float gg = tanhf(g[1024 + j]);
  float og = sigmoidf_(g[1536 + j]);
  float c2 = fg * c[(size_t)b * DD + j] + ig * gg;
  float h2 = og * tanhf(c2);
  c[(size_t)b * DD + j] = c2;
  h[(size_t)b * DD + j] = h2;
  unsigned short hb = f2bf(h2);
  xfull[(size_t)b * 1536 + 1024 + j] = hb;    // h slot of [emb|z|h]
  hz[(size_t)b * 1024 + j]           = hb;    // h2 slot of [h2|z]
}

// ---------------------------------------------------------------------------
extern "C" void kernel_launch(void* const* d_in, const int* in_sizes, int n_in,
                              void* d_out, int out_size, void* d_ws, size_t ws_size,
                              hipStream_t stream) {
  (void)in_sizes; (void)n_in; (void)out_size; (void)ws_size;
  const float* enc    = (const float*)d_in[0];
  const int*   caps   = (const int*)d_in[1];
  const int*   caplen = (const int*)d_in[2];
  const float* emb    = (const float*)d_in[3];
  const float* W_enc  = (const float*)d_in[4];  const float* b_enc  = (const float*)d_in[5];
  const float* W_dec  = (const float*)d_in[6];  const float* b_dec  = (const float*)d_in[7];
  const float* W_att  = (const float*)d_in[8];  const float* b_att  = (const float*)d_in[9];
  const float* W_ih   = (const float*)d_in[10]; const float* b_ih   = (const float*)d_in[11];
  const float* W_hh   = (const float*)d_in[12]; const float* b_hh   = (const float*)d_in[13];
  const float* W_h0   = (const float*)d_in[14]; const float* b_h0   = (const float*)d_in[15];
  const float* W_c0   = (const float*)d_in[16]; const float* b_c0   = (const float*)d_in[17];
  const float* W_beta = (const float*)d_in[18]; const float* b_beta = (const float*)d_in[19];
  const float* W_Lh   = (const float*)d_in[20]; const float* b_Lh   = (const float*)d_in[21];
  const float* W_Lz   = (const float*)d_in[22]; const float* b_Lz   = (const float*)d_in[23];
  const float* W_Lo   = (const float*)d_in[24]; const float* b_Lo   = (const float*)d_in[25];

  float* preds  = (float*)d_out;                          // (B, 19, V)
  float* alphas = preds + (size_t)BB * TM1 * VV;          // (B, 19, L)

  char* ws = (char*)d_ws; size_t off = 0;
  auto alloc = [&](size_t bytes) -> void* {
    void* p = ws + off; off = (off + bytes + 255) & ~(size_t)255; return p;
  };
  unsigned short* enc_b   = (unsigned short*)alloc((size_t)BB * LL * DD * 2);
  float*          enc_att = (float*)         alloc((size_t)BB * LL * DD * 4);
  unsigned short* Wenc_b  = (unsigned short*)alloc((size_t)DD * DD * 2);
  unsigned short* Wdec_b  = (unsigned short*)alloc((size_t)DD * DD * 2);
  unsigned short* Wh0_b   = (unsigned short*)alloc((size_t)DD * DD * 2);
  unsigned short* Wc0_b   = (unsigned short*)alloc((size_t)DD * DD * 2);
  unsigned short* Wg_b    = (unsigned short*)alloc((size_t)2048 * 1536 * 2);
  unsigned short* Wd_b    = (unsigned short*)alloc((size_t)DD * 1024 * 2);
  unsigned short* WLo_b   = (unsigned short*)alloc((size_t)VV * DD * 2);
  float*          bg      = (float*)         alloc(2048 * 4);
  float*          bd      = (float*)         alloc(DD * 4);
  unsigned short* mean_b  = (unsigned short*)alloc((size_t)BB * DD * 2);
  float*          h       = (float*)         alloc((size_t)BB * DD * 4);
  float*          c       = (float*)         alloc((size_t)BB * DD * 4);
  unsigned short* xfull   = (unsigned short*)alloc((size_t)BB * 1536 * 2);  // [emb|z|h]
  unsigned short* hz      = (unsigned short*)alloc((size_t)BB * 1024 * 2);  // [h2|z]
  float*          embf    = (float*)         alloc((size_t)BB * DD * 4);
  float*          dproj   = (float*)         alloc((size_t)BB * DD * 4);
  float*          betav   = (float*)         alloc(BB * 4);
  float*          zbuf    = (float*)         alloc((size_t)BB * DD * 4);
  float*          gates   = (float*)         alloc((size_t)BB * 2048 * 4);
  unsigned short* d_b     = (unsigned short*)alloc((size_t)BB * DD * 2);

  dim3 gblock(32, 4);
  auto ggrid = [](int N, int M) { return dim3((unsigned)((N + 255) / 256), (unsigned)(M / 32)); };
  auto lds_sz = [](int K) { return (size_t)(32 * K * 2); };  // A-tile staging bytes

  // ---- one-time prep ----
  cvt_kernel<<<512, 256, 0, stream>>>(W_enc, Wenc_b, (size_t)DD * DD);
  cvt_kernel<<<512, 256, 0, stream>>>(W_dec, Wdec_b, (size_t)DD * DD);
  cvt_kernel<<<512, 256, 0, stream>>>(W_h0,  Wh0_b,  (size_t)DD * DD);
  cvt_kernel<<<512, 256, 0, stream>>>(W_c0,  Wc0_b,  (size_t)DD * DD);
  cvt_kernel<<<2048, 256, 0, stream>>>(W_Lo, WLo_b,  (size_t)VV * DD);
  cvt_kernel<<<4096, 256, 0, stream>>>(enc,  enc_b,  (size_t)BB * LL * DD);
  pack2_kernel<<<2048, 256, 0, stream>>>(W_ih, 1024, W_hh, 512, Wg_b);
  pack2_kernel<<<512, 256, 0, stream>>>(W_Lh, 512, W_Lz, 512, Wd_b);
  addvec_kernel<<<8, 256, 0, stream>>>(b_ih, b_hh, bg, 2048);
  addvec_kernel<<<2, 256, 0, stream>>>(b_Lh, b_Lz, bd, DD);
  mean_kernel<<<BB, DD, 0, stream>>>(enc, mean_b);
  // h0 = tanh(mean @ W_h0^T + b): f32 h + bf16 into h-slot of xfull
  gemm_bf16_kernel<<<ggrid(DD, BB), gblock, lds_sz(DD), stream>>>(
      mean_b, DD, Wh0_b, DD, b_h0, nullptr, 0, h, DD, xfull + 1024, 1536, BB, DD, DD, 1);
  gemm_bf16_kernel<<<ggrid(DD, BB), gblock, lds_sz(DD), stream>>>(
      mean_b, DD, Wc0_b, DD, b_c0, nullptr, 0, c, DD, nullptr, 0, BB, DD, DD, 1);
  // enc_att = encoder_out @ W_enc^T + b_enc  (hoisted, 13.2 GFLOP)
  gemm_bf16_kernel<<<ggrid(DD, BB * LL), gblock, lds_sz(DD), stream>>>(
      enc_b, DD, Wenc_b, DD, b_enc, nullptr, 0, enc_att, DD, nullptr, 0, BB * LL, DD, DD, 0);

  // ---- 19 sequential decode steps ----
  for (int t = 0; t < TM1; ++t) {
    emb_kernel<<<BB, DD, 0, stream>>>(emb, caps, embf, xfull, t);
    // dec_proj = h @ W_dec^T + b_dec  (A = h-slot of xfull, lda=1536)
    gemm_bf16_kernel<<<ggrid(DD, BB), gblock, lds_sz(DD), stream>>>(
        xfull + 1024, 1536, Wdec_b, DD, b_dec, nullptr, 0, dproj, DD, nullptr, 0, BB, DD, DD, 0);
    beta_kernel<<<1, 128, 0, stream>>>(h, W_beta, b_beta, betav);
    attention_kernel<<<BB, 256, 0, stream>>>(enc_att, dproj, enc, betav, W_att, b_att,
                                             caplen, alphas, zbuf, xfull, hz, t);
    // gates = [emb|z|h] @ [W_ih|W_hh]^T + (b_ih + b_hh)
    gemm_bf16_kernel<<<ggrid(2048, BB), gblock, lds_sz(1536), stream>>>(
        xfull, 1536, Wg_b, 1536, bg, nullptr, 0, gates, 2048, nullptr, 0, BB, 2048, 1536, 0);
    lstm_kernel<<<BB, DD, 0, stream>>>(gates, h, c, xfull, hz);
    // d = emb + [h2|z] @ [W_Lh|W_Lz]^T + (b_Lh + b_Lz)  -> bf16
    gemm_bf16_kernel<<<ggrid(DD, BB), gblock, lds_sz(1024), stream>>>(
        hz, 1024, Wd_b, 1024, bd, embf, DD, nullptr, 0, d_b, DD, BB, DD, 1024, 0);
    // logits = d @ W_Lo^T + b_Lo -> preds[:, t, :]
    gemm_bf16_kernel<<<ggrid(VV, BB), gblock, lds_sz(DD), stream>>>(
        d_b, DD, WLo_b, DD, b_Lo, nullptr, 0, preds + (size_t)t * VV, TM1 * VV,
        nullptr, 0, BB, VV, DD, 0);
  }
}
